// _TemporalSSMBlock_83717502533973
// MI455X (gfx1250) — compile-verified, matrix operand order checked by reading, not run on hardware
//
#include <hip/hip_runtime.h>
#include <hip/hip_bf16.h>
#include <cstdint>

// Problem constants (from reference)
#define BB    2
#define LL    2048
#define DIMC  512
#define DIC   1024          // 2*DIM
#define NC    16
#define RC    32            // DIM/16
#define MROWS (BB*LL)       // 4096 token rows
#define CH    8             // scan chunks along L
#define CL    (LL/CH)       // 256 steps per chunk

typedef __attribute__((ext_vector_type(16))) _Float16 v16h;
typedef __attribute__((ext_vector_type(8)))  _Float16 v8h;
typedef __attribute__((ext_vector_type(8)))  float    v8f;

// ---------------------------------------------------------------------------
// CDNA5 async global->LDS copy (ASYNCcnt-tracked, no VGPR round trip).
// Generic LDS pointer's low 32 bits are the wave-relative LDS offset
// (flat aperture), which is what VDST expects.
// ---------------------------------------------------------------------------
__device__ __forceinline__ void async_copy_b128(const _Float16* g, _Float16* l) {
  unsigned lofs = (unsigned)(uintptr_t)l;
  asm volatile("global_load_async_to_lds_b128 %0, %1, off"
               :: "v"(lofs), "v"(g) : "memory");
}
__device__ __forceinline__ void async_wait0() {
  asm volatile("s_wait_asynccnt 0" ::: "memory");
}

// ---------------------------------------------------------------------------
// WMMA fragment: 16 rows x 32 K of f16 from row-major [row, K], leading dim
// ld. Per ISA 7.12.2: lane l -> row = l&15; elements 0..7 -> K = 8*(l>>4)+i,
// elements 8..15 -> K = 16 + 8*(l>>4) + i.  Two 128-bit loads.
// ---------------------------------------------------------------------------
__device__ __forceinline__ v16h load_frag16(const _Float16* __restrict__ base,
                                            int ld, int row, int k0, int hs) {
  const _Float16* p = base + (size_t)row * ld + k0 + 8 * hs;
  v8h lo = *(const v8h*)(p);
  v8h hi = *(const v8h*)(p + 16);
  v16h r;
#pragma unroll
  for (int i = 0; i < 8; ++i) { r[i] = lo[i]; r[i + 8] = hi[i]; }
  return r;
}

// Same fragment sourced from a 64x32 LDS tile (row-major, ld = 32).
__device__ __forceinline__ v16h lds_frag16(const _Float16* tile, int row, int hs) {
  const _Float16* p = tile + row * 32 + 8 * hs;
  v8h lo = *(const v8h*)(p);
  v8h hi = *(const v8h*)(p + 16);
  v16h r;
#pragma unroll
  for (int i = 0; i < 8; ++i) { r[i] = lo[i]; r[i + 8] = hi[i]; }
  return r;
}

// ---------------------------------------------------------------------------
// WMMA GEMM: C[m,n] = sum_k A[m,k]*B[n,k]   (A: MxK f16, B: NxK f16)
// Block = 8 waves = 128(M) x 64(N). The 64x32 B tile per K-step is shared by
// all 8 waves: staged into LDS with async global->LDS copies, double
// buffered; waves pull WMMA fragments from LDS (ds_load_b128). A fragments
// are per-wave and stay on the direct global path.
// MODE 0: plain store | 1: softplus(acc+ex[n]) | 2: acc + ex[m*N+n]
// Requires M%128==0, N%64==0, K%32==0 (true for all four GEMMs here).
// ---------------------------------------------------------------------------
template <int MODE>
__global__ __launch_bounds__(256)
void wmma_gemm_kernel(const _Float16* __restrict__ A,
                      const _Float16* __restrict__ Bm,
                      float* __restrict__ C,
                      int M, int N, int K,
                      const float* __restrict__ ex)
{
  __shared__ _Float16 bsh[2][64 * 32];          // 2 x 4 KB double buffer
  const int ntile = N >> 6;
  const int n0 = (blockIdx.x % ntile) << 6;
  const int m0 = ((blockIdx.x / ntile) << 7) + ((threadIdx.x >> 5) << 4);
  const int lane = threadIdx.x & 31;
  const int row = lane & 15;
  const int hs  = lane >> 4;

  // Cooperative B staging: thread t moves one b128 (row = t>>2, seg = t&3).
  const int brow = threadIdx.x >> 2;
  const int seg  = threadIdx.x & 3;
  const _Float16* bsrc = Bm + (size_t)(n0 + brow) * K + seg * 8;
  _Float16* bdst = &bsh[0][0] + brow * 32 + seg * 8;

  const int nk = K >> 5;
  async_copy_b128(bsrc, bdst);                  // prologue: tile 0

  v8f acc0 = {}, acc1 = {}, acc2 = {}, acc3 = {};
  for (int t = 0; t < nk; ++t) {
    async_wait0();                              // my staged b128 is in LDS
    __syncthreads();                            // everyone's is
    if (t + 1 < nk)                             // stage next tile (other buf)
      async_copy_b128(bsrc + (t + 1) * 32,
                      &bsh[(t + 1) & 1][0] + brow * 32 + seg * 8);
    if (t + 1 < nk)
      __builtin_prefetch(A + (size_t)(m0 + row) * K + (t + 1) * 32, 0, 0);

    v16h a  = load_frag16(A, K, m0 + row, t * 32, hs);
    const _Float16* bb = &bsh[t & 1][0];
    v16h b0 = lds_frag16(bb,           row, hs);
    v16h b1 = lds_frag16(bb + 16 * 32, row, hs);
    v16h b2 = lds_frag16(bb + 32 * 32, row, hs);
    v16h b3 = lds_frag16(bb + 48 * 32, row, hs);
    acc0 = __builtin_amdgcn_wmma_f32_16x16x32_f16(false, a, false, b0, (short)0, acc0, false, false);
    acc1 = __builtin_amdgcn_wmma_f32_16x16x32_f16(false, a, false, b1, (short)0, acc1, false, false);
    acc2 = __builtin_amdgcn_wmma_f32_16x16x32_f16(false, a, false, b2, (short)0, acc2, false, false);
    acc3 = __builtin_amdgcn_wmma_f32_16x16x32_f16(false, a, false, b3, (short)0, acc3, false, false);
  }

  v8f accs[4] = {acc0, acc1, acc2, acc3};
#pragma unroll
  for (int t = 0; t < 4; ++t) {
    const int n = n0 + 16 * t + row;
#pragma unroll
    for (int r = 0; r < 8; ++r) {
      const int m = m0 + r + 8 * hs;   // C/D layout: M = vgpr + 8*(lane>>4)
      float v = accs[t][r];
      if (MODE == 1) {
        v += ex[n];
        v = (v > 20.f) ? v : log1pf(__expf(v));       // stable softplus
      } else if (MODE == 2) {
        v += ex[(size_t)m * N + n];
      }
      C[(size_t)m * N + n] = v;
    }
  }
}

// ---------------------------------------------------------------------------
// f32 -> f16 converter (weights, once per call)
// ---------------------------------------------------------------------------
__global__ void f32_to_f16_kernel(const float* __restrict__ in,
                                  _Float16* __restrict__ outp, int n) {
  int i = blockIdx.x * blockDim.x + threadIdx.x;
  if (i < n) outp[i] = (_Float16)in[i];
}

// ---------------------------------------------------------------------------
// LayerNorm over DIM=512, one block (256 thr) per token row; writes f16 h.
// ---------------------------------------------------------------------------
__global__ __launch_bounds__(256)
void ln_kernel(const float* __restrict__ x, const float* __restrict__ g,
               const float* __restrict__ be, _Float16* __restrict__ h16)
{
  const int rowid = blockIdx.x;
  const float* xr = x + (size_t)rowid * DIMC;
  const int tid = threadIdx.x;
  float v0 = xr[tid], v1 = xr[tid + 256];
  float s = v0 + v1, ss = v0 * v0 + v1 * v1;
#pragma unroll
  for (int off = 16; off >= 1; off >>= 1) {
    s  += __shfl_xor(s,  off, 32);
    ss += __shfl_xor(ss, off, 32);
  }
  __shared__ float rs[8], rss[8];
  if ((tid & 31) == 0) { rs[tid >> 5] = s; rss[tid >> 5] = ss; }
  __syncthreads();
  float ts = 0.f, tss = 0.f;
#pragma unroll
  for (int i = 0; i < 8; ++i) { ts += rs[i]; tss += rss[i]; }
  const float mu  = ts * (1.f / DIMC);
  const float var = tss * (1.f / DIMC) - mu * mu;
  const float inv = rsqrtf(var + 1e-5f);
  _Float16* hr = h16 + (size_t)rowid * DIMC;
  hr[tid]       = (_Float16)((v0 - mu) * inv * g[tid]       + be[tid]);
  hr[tid + 256] = (_Float16)((v1 - mu) * inv * g[tid + 256] + be[tid + 256]);
}

// ---------------------------------------------------------------------------
// Depthwise causal conv (4 taps) + bias + SiLU over x_in = xz[..., :DI].
// ---------------------------------------------------------------------------
__global__ __launch_bounds__(256)
void conv_silu_kernel(const float* __restrict__ xz,
                      const float* __restrict__ w_conv,
                      const float* __restrict__ b_conv,
                      _Float16* __restrict__ xa16)
{
  int idx = blockIdx.x * blockDim.x + threadIdx.x;
  if (idx >= BB * LL * DIC) return;
  const int d = idx & (DIC - 1);
  const int l = (idx >> 10) & (LL - 1);
  const int b = idx >> 21;
  float acc = b_conv[d];
#pragma unroll
  for (int t = 0; t < 4; ++t) {
    const int ls = l - 3 + t;
    if (ls >= 0)
      acc += w_conv[d * 4 + t] * xz[((size_t)(b * LL + ls)) * (2 * DIC) + d];
  }
  const float sv = acc / (1.f + __expf(-acc));   // SiLU
  xa16[idx] = (_Float16)sv;
}

// Extract dt_low (first R cols of x_dbl) as f16 for the dt GEMM.
__global__ void dtlow_kernel(const float* __restrict__ xdbl,
                             _Float16* __restrict__ dtlow16) {
  int i = blockIdx.x * blockDim.x + threadIdx.x;
  if (i >= MROWS * RC) return;
  const int m = i >> 5, r = i & 31;
  dtlow16[i] = (_Float16)xdbl[(size_t)m * 64 + r];
}

// ---------------------------------------------------------------------------
// Chunk-parallel selective scan.  The recurrence s <- a*s + b is linear, so
// each chunk of CL=256 steps reduces to a transfer pair (prod a, combined b).
// Pass 1: per-(b,d,n,chunk) transfer pairs          (8192 waves -> latency ok)
// Pass 2: tiny sequential prefix over CH=8 chunks   (per (b,d,n))
// Pass 3: re-run each chunk from its incoming state, contract with C, add
//         D-skip, gate with SiLU(z), write f16 y.   (8192 waves)
// ---------------------------------------------------------------------------
__global__ __launch_bounds__(256)
void scan_pass1(const float* __restrict__ dt, const _Float16* __restrict__ xa16,
                const float* __restrict__ xdbl, const float* __restrict__ A_log,
                float* __restrict__ Aprod, float* __restrict__ Bacc)
{
  int idx = blockIdx.x * blockDim.x + threadIdx.x;
  if (idx >= BB * DIC * CH * NC) return;
  const int n  = idx & (NC - 1);
  const int ch = (idx >> 4) & (CH - 1);
  const int d  = (idx >> 7) & (DIC - 1);
  const int b  = idx >> 17;
  const float Ad = -__expf(A_log[d * NC + n]);
  float ap = 1.f, bc = 0.f;
  const int l0 = ch * CL;
  for (int i = 0; i < CL; ++i) {
    const size_t rowb = (size_t)(b * LL + l0 + i);
    const float dtv = dt[rowb * DIC + d];
    const float xav = (float)xa16[rowb * DIC + d];
    const float Bv  = xdbl[rowb * 64 + RC + n];
    const float dA  = __expf(dtv * Ad);
    ap *= dA;
    bc = bc * dA + dtv * Bv * xav;
  }
  Aprod[idx] = ap;
  Bacc[idx]  = bc;
}

__global__ __launch_bounds__(256)
void scan_pass2(const float* __restrict__ Aprod, const float* __restrict__ Bacc,
                float* __restrict__ Sinit)
{
  int idx = blockIdx.x * blockDim.x + threadIdx.x;
  if (idx >= BB * DIC * NC) return;
  const int n = idx & (NC - 1);
  const int bd = idx >> 4;                 // b*DIC + d
  const size_t base = ((size_t)bd * CH) * NC + n;
  float s = 0.f;
#pragma unroll
  for (int c = 0; c < CH; ++c) {
    Sinit[base + (size_t)c * NC] = s;
    s = Aprod[base + (size_t)c * NC] * s + Bacc[base + (size_t)c * NC];
  }
}

__global__ __launch_bounds__(256)
void scan_pass3(const float* __restrict__ dt, const _Float16* __restrict__ xa16,
                const float* __restrict__ xdbl, const float* __restrict__ xz,
                const float* __restrict__ A_log, const float* __restrict__ Dp,
                const float* __restrict__ Sinit, _Float16* __restrict__ y16)
{
  int idx = blockIdx.x * blockDim.x + threadIdx.x;
  if (idx >= BB * DIC * CH * NC) return;
  const int n  = idx & (NC - 1);
  const int ch = (idx >> 4) & (CH - 1);
  const int d  = (idx >> 7) & (DIC - 1);
  const int b  = idx >> 17;
  const float Ad = -__expf(A_log[d * NC + n]);
  const float Dd = Dp[d];
  float state = Sinit[idx];
  const int l0 = ch * CL;
  for (int i = 0; i < CL; ++i) {
    const size_t rowb = (size_t)(b * LL + l0 + i);
    const float dtv = dt[rowb * DIC + d];
    const float xav = (float)xa16[rowb * DIC + d];
    const float Bv  = xdbl[rowb * 64 + RC + n];
    const float Cv  = xdbl[rowb * 64 + RC + NC + n];
    const float dA  = __expf(dtv * Ad);
    state = state * dA + dtv * Bv * xav;
    float c = state * Cv;
#pragma unroll
    for (int off = 8; off >= 1; off >>= 1) c += __shfl_xor(c, off, 16);
    if (n == 0) {
      const float zv = xz[rowb * (2 * DIC) + DIC + d];
      const float yv = (c + xav * Dd) * (zv / (1.f + __expf(-zv)));
      y16[rowb * DIC + d] = (_Float16)yv;
    }
  }
}

// ---------------------------------------------------------------------------
// Host-side orchestration
// ---------------------------------------------------------------------------
static inline int gemm_grid(int M, int N) { return (M / 128) * (N / 64); }
static inline int cdiv(long a, long b) { return (int)((a + b - 1) / b); }

extern "C" void kernel_launch(void* const* d_in, const int* in_sizes, int n_in,
                              void* d_out, int out_size, void* d_ws, size_t ws_size,
                              hipStream_t stream)
{
  (void)in_sizes; (void)n_in; (void)out_size; (void)ws_size;
  const float* x       = (const float*)d_in[0];
  const float* ln_g    = (const float*)d_in[1];
  const float* ln_b    = (const float*)d_in[2];
  const float* w_in    = (const float*)d_in[3];
  const float* w_conv  = (const float*)d_in[4];
  const float* b_conv  = (const float*)d_in[5];
  const float* w_xproj = (const float*)d_in[6];
  const float* w_dt    = (const float*)d_in[7];
  const float* b_dt    = (const float*)d_in[8];
  const float* A_log   = (const float*)d_in[9];
  const float* Dp      = (const float*)d_in[10];
  const float* w_out   = (const float*)d_in[11];
  float* out = (float*)d_out;

  // Workspace carve-up (~79 MB total)
  char* ws = (char*)d_ws;
  size_t off = 0;
  auto carve = [&](size_t bytes) -> void* {
    void* p = ws + off;
    off += (bytes + 255) & ~(size_t)255;
    return p;
  };
  _Float16* w_in16    = (_Float16*)carve((size_t)2 * DIC * DIMC * 2);   // 2 MB
  _Float16* w_xproj16 = (_Float16*)carve((size_t)64 * DIC * 2);         // 128 KB
  _Float16* w_dt16    = (_Float16*)carve((size_t)DIC * RC * 2);         // 64 KB
  _Float16* w_out16   = (_Float16*)carve((size_t)DIMC * DIC * 2);       // 1 MB
  _Float16* h16       = (_Float16*)carve((size_t)MROWS * DIMC * 2);     // 4 MB
  float*    xz        = (float*)   carve((size_t)MROWS * 2 * DIC * 4);  // 32 MB
  _Float16* xa16      = (_Float16*)carve((size_t)MROWS * DIC * 2);      // 8 MB
  float*    xdbl      = (float*)   carve((size_t)MROWS * 64 * 4);       // 1 MB
  _Float16* dtlow16   = (_Float16*)carve((size_t)MROWS * RC * 2);       // 256 KB
  float*    dtf       = (float*)   carve((size_t)MROWS * DIC * 4);      // 16 MB
  _Float16* y16       = (_Float16*)carve((size_t)MROWS * DIC * 2);      // 8 MB
  float*    aprod     = (float*)   carve((size_t)BB * DIC * CH * NC * 4); // 1 MB
  float*    bacc      = (float*)   carve((size_t)BB * DIC * CH * NC * 4); // 1 MB
  float*    sinit     = (float*)   carve((size_t)BB * DIC * CH * NC * 4); // 1 MB

  dim3 blk(256);

  // Weight conversions (f32 -> f16), once per call
  f32_to_f16_kernel<<<cdiv((long)2 * DIC * DIMC, 256), blk, 0, stream>>>(w_in, w_in16, 2 * DIC * DIMC);
  f32_to_f16_kernel<<<cdiv((long)64 * DIC, 256), blk, 0, stream>>>(w_xproj, w_xproj16, 64 * DIC);
  f32_to_f16_kernel<<<cdiv((long)DIC * RC, 256), blk, 0, stream>>>(w_dt, w_dt16, DIC * RC);
  f32_to_f16_kernel<<<cdiv((long)DIMC * DIC, 256), blk, 0, stream>>>(w_out, w_out16, DIMC * DIC);

  // 1. LayerNorm -> h16
  ln_kernel<<<MROWS, blk, 0, stream>>>(x, ln_g, ln_b, h16);

  // 2. xz = h @ w_in^T   (4096 x 2048 x 512)
  wmma_gemm_kernel<0><<<gemm_grid(MROWS, 2 * DIC), blk, 0, stream>>>(
      h16, w_in16, xz, MROWS, 2 * DIC, DIMC, nullptr);

  // 3. depthwise conv + SiLU -> xa16
  conv_silu_kernel<<<cdiv((long)BB * LL * DIC, 256), blk, 0, stream>>>(
      xz, w_conv, b_conv, xa16);

  // 4. x_dbl = x_a @ w_xproj^T  (4096 x 64 x 1024)
  wmma_gemm_kernel<0><<<gemm_grid(MROWS, 64), blk, 0, stream>>>(
      xa16, w_xproj16, xdbl, MROWS, 64, DIC, nullptr);

  // 5. dt_low extract (f16)
  dtlow_kernel<<<cdiv((long)MROWS * RC, 256), blk, 0, stream>>>(xdbl, dtlow16);

  // 6. dt = softplus(dt_low @ w_dt^T + b_dt)  (4096 x 1024 x 32)
  wmma_gemm_kernel<1><<<gemm_grid(MROWS, DIC), blk, 0, stream>>>(
      dtlow16, w_dt16, dtf, MROWS, DIC, RC, b_dt);

  // 7. chunk-parallel selective scan (3 passes), fused gate -> y16
  scan_pass1<<<cdiv((long)BB * DIC * CH * NC, 256), blk, 0, stream>>>(
      dtf, xa16, xdbl, A_log, aprod, bacc);
  scan_pass2<<<cdiv((long)BB * DIC * NC, 256), blk, 0, stream>>>(
      aprod, bacc, sinit);
  scan_pass3<<<cdiv((long)BB * DIC * CH * NC, 256), blk, 0, stream>>>(
      dtf, xa16, xdbl, xz, A_log, Dp, sinit, y16);

  // 8. out = x + y @ w_out^T  (4096 x 512 x 1024)
  wmma_gemm_kernel<2><<<gemm_grid(MROWS, DIMC), blk, 0, stream>>>(
      y16, w_out16, out, MROWS, DIMC, DIC, x);
}
